// VSSBlock_zorder_local_shift_42812234006789
// MI455X (gfx1250) — matrix operand order repaired and checked by reading, not compiled, not run on hardware
//
#include <hip/hip_runtime.h>
#include <hip/hip_bf16.h>

typedef _Float16 v16h __attribute__((ext_vector_type(16)));
typedef _Float16 hvec8 __attribute__((ext_vector_type(8)));
typedef float    v8f  __attribute__((ext_vector_type(8)));

// ---------------- problem dimensions ----------------
#define BATCH   2
#define CCH     128
#define HH      128
#define WW      128
#define NPIX    (BATCH*HH*WW)          // 32768
#define DIN     256                    // d_inner
#define NST     16                     // d_state
#define DTR     8                      // dt_rank
#define KDIR    4
#define NWIN    512                    // BATCH*(128/8)*(128/8)
#define WTOK    64
#define C3      42                     // C//3
#define C3P     48                     // padded to 16
#define XDBLC   48                     // dt_rank + 2*d_state = 40 -> pad 48
#define K1      1152                   // 128*9
#define K2P     384                    // 42*9=378 -> pad 384
#define CA      4                      // C//30

// ---------------- workspace layout ----------------
static constexpr size_t al256(size_t x){ return (x + 255) & ~(size_t)255; }
static constexpr size_t O_WB_IN  = 0;                                   // 512x128 f16
static constexpr size_t O_WB_OUT = al256(O_WB_IN  + (size_t)512*128*2); // 128x256 f16
static constexpr size_t O_WB_XP  = al256(O_WB_OUT + (size_t)128*256*2); // 4x48x256 f16
static constexpr size_t O_WB_C1  = al256(O_WB_XP  + (size_t)4*48*256*2);// 48x1152 f16
static constexpr size_t O_WB_C2  = al256(O_WB_C1  + (size_t)48*1152*2); // 128x384 f16
static constexpr size_t O_XN     = al256(O_WB_C2  + (size_t)128*384*2); // 32768x128 f16
static constexpr size_t O_XZ     = al256(O_XN   + (size_t)NPIX*128*2);  // 32768x512 f32
static constexpr size_t O_XC     = al256(O_XZ   + (size_t)NPIX*512*4);  // 32768x256 f32
static constexpr size_t O_XS     = al256(O_XC   + (size_t)NPIX*256*4);  // 4x512x64x256 f16
static constexpr size_t O_XDBL   = al256(O_XS   + (size_t)KDIR*NWIN*WTOK*DIN*2); // 4x32768x48 f32
static constexpr size_t O_DELTA  = al256(O_XDBL + (size_t)KDIR*NPIX*XDBLC*4);    // 4x32768x256 f32
static constexpr size_t O_YS     = al256(O_DELTA+ (size_t)KDIR*NPIX*DIN*4);      // 4x32768x256 f32
static constexpr size_t O_YG     = al256(O_YS   + (size_t)KDIR*NPIX*DIN*4);      // 32768x256 f16
static constexpr size_t O_YP     = al256(O_YG   + (size_t)NPIX*DIN*2);           // 32768x128 f32
static constexpr size_t O_X1     = al256(O_YP   + (size_t)NPIX*128*4);           // 32768x128 f32
static constexpr size_t O_TN     = al256(O_X1   + (size_t)NPIX*128*4);           // 32768x128 f16
static constexpr size_t O_A1     = al256(O_TN   + (size_t)NPIX*128*2);           // 32768x1152 f16
static constexpr size_t O_T1     = al256(O_A1   + (size_t)NPIX*K1*2);            // 32768x48 f32
static constexpr size_t O_A2     = al256(O_T1   + (size_t)NPIX*C3P*4);           // 32768x384 f16
static constexpr size_t O_T2     = al256(O_A2   + (size_t)NPIX*K2P*2);           // 32768x128 f32
static constexpr size_t O_P      = al256(O_T2   + (size_t)NPIX*128*4);           // 2x128 f32
static constexpr size_t O_ATTN   = al256(O_P    + (size_t)256*4);                // 2x128 f32

// ---------------- device helpers ----------------
__device__ __forceinline__ float sigmoidf_(float v){ return 1.f/(1.f + expf(-v)); }
__device__ __forceinline__ float siluf_(float v){ return v * sigmoidf_(v); }
__device__ __forceinline__ float softplusf_(float v){ return v > 20.f ? v : log1pf(expf(v)); }
__device__ __forceinline__ float geluf_(float v){ return 0.5f*v*(1.f + erff(v*0.70710678118654752f)); }

// z-order: P0[t] = i*8+j where i = odd bits of t, j = even bits of t
__device__ __forceinline__ void zdecode(int t, int& i, int& j){
  j = (t & 1) | (((t >> 2) & 1) << 1) | (((t >> 4) & 1) << 2);
  i = ((t >> 1) & 1) | (((t >> 3) & 1) << 1) | (((t >> 5) & 1) << 2);
}
// INV0[i*8+j] = morton(i,j) (i in odd bits)
__device__ __forceinline__ int mortonIJ(int i, int j){
  int t = 0;
  #pragma unroll
  for (int b = 0; b < 3; ++b){ t |= ((j >> b) & 1) << (2*b); t |= ((i >> b) & 1) << (2*b + 1); }
  return t;
}

// ---------------- weight prep (f32 -> f16, with padding) ----------------
__global__ void k_prep_inproj(const float* __restrict__ w, _Float16* __restrict__ o){
  int i = blockIdx.x*256 + threadIdx.x;
  if (i < 512*128) o[i] = (_Float16)w[i];
}
__global__ void k_prep_outproj(const float* __restrict__ w, _Float16* __restrict__ o){
  int i = blockIdx.x*256 + threadIdx.x;
  if (i < 128*256) o[i] = (_Float16)w[i];
}
__global__ void k_prep_xproj(const float* __restrict__ w, _Float16* __restrict__ o){
  int i = blockIdx.x*256 + threadIdx.x;
  if (i >= 4*48*256) return;
  int k = i / (48*256), r = i % (48*256), n = r / 256, kk = r % 256;
  o[i] = (n < 40) ? (_Float16)w[((size_t)k*40 + n)*256 + kk] : (_Float16)0.f;
}
__global__ void k_prep_c1(const float* __restrict__ w, _Float16* __restrict__ o){
  int i = blockIdx.x*256 + threadIdx.x;
  if (i >= 48*K1) return;
  int oc = i / K1, r = i % K1, tap = r / 128, ic = r % 128;
  o[i] = (oc < C3) ? (_Float16)w[((size_t)oc*128 + ic)*9 + tap] : (_Float16)0.f;
}
__global__ void k_prep_c2(const float* __restrict__ w, _Float16* __restrict__ o){
  int i = blockIdx.x*256 + threadIdx.x;
  if (i >= 128*K2P) return;
  int oc = i / K2P, r = i % K2P;
  if (r < 378){ int tap = r / C3, ic = r % C3; o[i] = (_Float16)w[((size_t)oc*C3 + ic)*9 + tap]; }
  else o[i] = (_Float16)0.f;
}

// ---------------- generic WMMA GEMM: C[M,N] = A[M,K](f16) * B[N,K](f16)^T ----------------
// block = 64 threads (2 waves), each wave computes one 16x16 tile (stacked in M).
__global__ void k_gemm_wmma(const _Float16* __restrict__ A, const _Float16* __restrict__ B,
                            float* __restrict__ C, int M, int N, int K,
                            long aBatch, long bBatch, long cBatch){
  const _Float16* Ab = A + (long)blockIdx.z * aBatch;
  const _Float16* Bb = B + (long)blockIdx.z * bBatch;
  float*          Cb = C + (long)blockIdx.z * cBatch;
  const int wave = threadIdx.x >> 5;
  const int lane = threadIdx.x & 31;
  const int tm = (blockIdx.y * 2 + wave) * 16;
  const int tn = blockIdx.x * 16;
  if (tm >= M) return;
  const int hlf = lane >> 4;       // 0/1
  const int l15 = lane & 15;
  v8f acc = {};
  const _Float16* arow = Ab + (long)(tm + l15) * K + hlf * 8;   // A fragment rows
  const _Float16* brow = Bb + (long)(tn + l15) * K + hlf * 8;   // B fragment cols
  for (int k0 = 0; k0 < K; k0 += 32){
    hvec8 alo = *(const hvec8*)(arow + k0);
    hvec8 ahi = *(const hvec8*)(arow + k0 + 16);
    hvec8 blo = *(const hvec8*)(brow + k0);
    hvec8 bhi = *(const hvec8*)(brow + k0 + 16);
    v16h a, b;
    #pragma unroll
    for (int e = 0; e < 8; ++e){ a[e]=alo[e]; a[e+8]=ahi[e]; b[e]=blo[e]; b[e+8]=bhi[e]; }
    __builtin_prefetch(arow + k0 + 128, 0, 0);   // global_prefetch_b8
    acc = __builtin_amdgcn_wmma_f32_16x16x32_f16(false, a, false, b, (short)0, acc, false, false);
  }
  const int mrow = tm + hlf * 8;
  #pragma unroll
  for (int r = 0; r < 8; ++r) Cb[(long)(mrow + r) * N + tn + l15] = acc[r];
}

// ---------------- LN1 + roll(-4,-4), output f16 [pix][128] ----------------
__global__ void k_ln1_shift(const float* __restrict__ x, const float* __restrict__ w,
                            const float* __restrict__ b, _Float16* __restrict__ xn){
  const int pix = blockIdx.x, c = threadIdx.x;
  const int bi = pix >> 14, y = (pix >> 7) & 127, xx = pix & 127;
  const int ysrc = (y + 4) & 127, xsrc = (xx + 4) & 127;
  float v = x[(((size_t)(bi*128 + c)) << 14) + (ysrc << 7) + xsrc];
  __shared__ float sb[128];
  sb[c] = v; __syncthreads();
  for (int s = 64; s > 0; s >>= 1){ if (c < s) sb[c] += sb[c+s]; __syncthreads(); }
  float mean = sb[0] * (1.f/128.f); __syncthreads();
  float d = v - mean; sb[c] = d*d; __syncthreads();
  for (int s = 64; s > 0; s >>= 1){ if (c < s) sb[c] += sb[c+s]; __syncthreads(); }
  float rstd = rsqrtf(sb[0]*(1.f/128.f) + 1e-5f);
  xn[(size_t)pix*128 + c] = (_Float16)(d * rstd * w[c] + b[c]);
}

// ---------------- depthwise 3x3 + SiLU on xi (= first 256 cols of XZ) ----------------
__global__ void k_dwconv_silu(const float* __restrict__ xz, const float* __restrict__ cw,
                              const float* __restrict__ cb, float* __restrict__ xc){
  const int pix = blockIdx.x, d = threadIdx.x;
  const int bi = pix >> 14, y = (pix >> 7) & 127, xx = pix & 127;
  float acc = cb[d];
  #pragma unroll
  for (int ky = 0; ky < 3; ++ky){
    int yy = y + ky - 1; if ((unsigned)yy >= 128u) continue;
    #pragma unroll
    for (int kx = 0; kx < 3; ++kx){
      int xx2 = xx + kx - 1; if ((unsigned)xx2 >= 128u) continue;
      size_t np = ((size_t)bi << 14) + (yy << 7) + xx2;
      acc += xz[np*512 + d] * cw[d*9 + ky*3 + kx];
    }
  }
  xc[(size_t)pix*256 + d] = siluf_(acc);
}

// ---------------- gather 4-direction z-order sequences (+ shift mask), f16 ----------------
__global__ void k_gather_xs(const float* __restrict__ xc, _Float16* __restrict__ xs){
  const int wwin = blockIdx.x, k = blockIdx.y, l = blockIdx.z, d = threadIdx.x;
  int t = (k >= 2) ? (63 - l) : l;
  int i, j; zdecode(t, i, j);
  int py = (k & 1) ? j : i;          // P1 swaps i/j
  int px = (k & 1) ? i : j;
  const int bi = wwin >> 8, wy = (wwin >> 4) & 15, wx = wwin & 15;
  const int y = wy*8 + py, x = wx*8 + px;
  float v = 0.f;
  if (y < 124 && x < 124){
    size_t pix = ((size_t)bi << 14) + (y << 7) + x;
    v = xc[pix*256 + d];
  }
  xs[(((size_t)k*NWIN + wwin)*WTOK + l)*DIN + d] = (_Float16)v;
}

// ---------------- delta = softplus(dts @ dt_proj_w^T + b) ----------------
__global__ void k_delta(const float* __restrict__ xdbl, const float* __restrict__ dtw,
                        const float* __restrict__ dtb, float* __restrict__ delta){
  const int m = blockIdx.x, k = blockIdx.y, d = threadIdx.x;
  __shared__ float sdts[DTR];
  if (d < DTR) sdts[d] = xdbl[((size_t)k*NPIX + m)*XDBLC + d];
  __syncthreads();
  float acc = dtb[k*DIN + d];
  #pragma unroll
  for (int r = 0; r < DTR; ++r) acc += sdts[r] * dtw[((size_t)k*DIN + d)*DTR + r];
  delta[((size_t)k*NPIX + m)*DIN + d] = softplusf_(acc);
}

// ---------------- selective scan: one block per (window, direction) ----------------
__global__ void k_scan(const float* __restrict__ xdbl, const float* __restrict__ delta,
                       const _Float16* __restrict__ xs, const float* __restrict__ alog,
                       const float* __restrict__ dsv, float* __restrict__ ys){
  const int wwin = blockIdx.x, k = blockIdx.y, d = threadIdx.x;
  __shared__ float sB[WTOK][NST];
  __shared__ float sC[WTOK][NST];
  for (int e = d; e < WTOK*NST; e += 256){
    int l = e >> 4, n = e & 15;
    size_t row = ((size_t)k*NPIX + wwin*WTOK + l)*XDBLC;
    sB[l][n] = xdbl[row + 8 + n];
    sC[l][n] = xdbl[row + 24 + n];
  }
  __syncthreads();
  float A[NST], h[NST];
  #pragma unroll
  for (int n = 0; n < NST; ++n){
    A[n] = -expf(alog[((size_t)k*DIN + d)*NST + n]);
    h[n] = 0.f;
  }
  const float Dv = dsv[k*DIN + d];
  for (int l = 0; l < WTOK; ++l){
    size_t m = (size_t)wwin*WTOK + l;
    float dt = delta[((size_t)k*NPIX + m)*DIN + d];
    float xv = (float)xs[(((size_t)k*NWIN + wwin)*WTOK + l)*DIN + d];
    float dx = dt * xv, acc = 0.f;
    #pragma unroll
    for (int n = 0; n < NST; ++n){
      h[n] = h[n] * expf(dt * A[n]) + dx * sB[l][n];
      acc += h[n] * sC[l][n];
    }
    ys[((size_t)k*NPIX + m)*DIN + d] = acc + xv * Dv;
  }
}

// ---------------- combine 4 dirs + window_reverse + out_norm + SiLU(z) gate ----------------
__global__ void k_combine_gate(const float* __restrict__ ys, const float* __restrict__ xz,
                               const float* __restrict__ onw, const float* __restrict__ onb,
                               _Float16* __restrict__ yg){
  const int wwin = blockIdx.x, l = blockIdx.y, d = threadIdx.x;
  const int py = l >> 3, px = l & 7;
  const int t0 = mortonIJ(py, px), t1 = mortonIJ(px, py);
  const size_t base = (size_t)wwin*WTOK;
  float v = ys[((size_t)0*NPIX + base + t0)*DIN + d]
          + ys[((size_t)1*NPIX + base + t1)*DIN + d]
          + ys[((size_t)2*NPIX + base + (63 - t0))*DIN + d]
          + ys[((size_t)3*NPIX + base + (63 - t1))*DIN + d];
  __shared__ float sb[256];
  sb[d] = v; __syncthreads();
  for (int s = 128; s > 0; s >>= 1){ if (d < s) sb[d] += sb[d+s]; __syncthreads(); }
  float mean = sb[0] * (1.f/256.f); __syncthreads();
  float df = v - mean; sb[d] = df*df; __syncthreads();
  for (int s = 128; s > 0; s >>= 1){ if (d < s) sb[d] += sb[d+s]; __syncthreads(); }
  float rstd = rsqrtf(sb[0]*(1.f/256.f) + 1e-5f);
  const int bi = wwin >> 8, wy = (wwin >> 4) & 15, wx = wwin & 15;
  const size_t pix = ((size_t)bi << 14) + ((wy*8 + py) << 7) + (wx*8 + px);
  float z = xz[pix*512 + 256 + d];
  yg[pix*256 + d] = (_Float16)((df * rstd * onw[d] + onb[d]) * siluf_(z));
}

// ---------------- roll back + residual: x1 = shortcut*skip + y ----------------
__global__ void k_x1(const float* __restrict__ x, const float* __restrict__ yp,
                     const float* __restrict__ skip, float* __restrict__ x1){
  const int pix = blockIdx.x, c = threadIdx.x;
  const int bi = pix >> 14, y = (pix >> 7) & 127, xx = pix & 127;
  const size_t rpix = ((size_t)bi << 14) + (((y - 4) & 127) << 7) + ((xx - 4) & 127);
  float sc = x[(((size_t)(bi*128 + c)) << 14) + (y << 7) + xx];
  x1[(size_t)pix*128 + c] = sc * skip[c] + yp[rpix*128 + c];
}

// ---------------- LN2 -> f16 ----------------
__global__ void k_ln2(const float* __restrict__ x1, const float* __restrict__ w,
                      const float* __restrict__ b, _Float16* __restrict__ tn){
  const int pix = blockIdx.x, c = threadIdx.x;
  float v = x1[(size_t)pix*128 + c];
  __shared__ float sb[128];
  sb[c] = v; __syncthreads();
  for (int s = 64; s > 0; s >>= 1){ if (c < s) sb[c] += sb[c+s]; __syncthreads(); }
  float mean = sb[0] * (1.f/128.f); __syncthreads();
  float d = v - mean; sb[c] = d*d; __syncthreads();
  for (int s = 64; s > 0; s >>= 1){ if (c < s) sb[c] += sb[c+s]; __syncthreads(); }
  float rstd = rsqrtf(sb[0]*(1.f/128.f) + 1e-5f);
  tn[(size_t)pix*128 + c] = (_Float16)(d * rstd * w[c] + b[c]);
}

// ---------------- im2col for conv1 (K index = tap*128 + c) ----------------
__global__ void k_im2col1(const _Float16* __restrict__ tn, _Float16* __restrict__ a1){
  const int pix = blockIdx.x, tap = blockIdx.y, c = threadIdx.x;
  const int bi = pix >> 14, y = (pix >> 7) & 127, xx = pix & 127;
  const int yy = y + tap/3 - 1, xx2 = xx + tap%3 - 1;
  _Float16 v = (_Float16)0.f;
  if ((unsigned)yy < 128u && (unsigned)xx2 < 128u)
    v = tn[(((size_t)bi << 14) + (yy << 7) + xx2)*128 + c];
  a1[(size_t)pix*K1 + tap*128 + c] = v;
}

// ---------------- gelu(conv1+b) + im2col for conv2 (K index = tap*42 + i) ----------------
__global__ void k_gelu_im2col2(const float* __restrict__ t1, const float* __restrict__ c1b,
                               _Float16* __restrict__ a2){
  const int pix = blockIdx.x, tap = blockIdx.y, i = threadIdx.x;
  const int bi = pix >> 14, y = (pix >> 7) & 127, xx = pix & 127;
  if (i < C3){
    const int yy = y + tap/3 - 1, xx2 = xx + tap%3 - 1;
    float v = 0.f;
    if ((unsigned)yy < 128u && (unsigned)xx2 < 128u){
      size_t p2 = ((size_t)bi << 14) + (yy << 7) + xx2;
      v = geluf_(t1[p2*C3P + i] + c1b[i]);
    }
    a2[(size_t)pix*K2P + tap*C3 + i] = (_Float16)v;
  } else if (tap == 0){
    a2[(size_t)pix*K2P + 378 + (i - C3)] = (_Float16)0.f;  // zero tail pad
  }
}

// ---------------- global mean pool of (t2 + c2_b) ----------------
__global__ void k_pool(const float* __restrict__ t2, const float* __restrict__ c2b,
                       float* __restrict__ p){
  const int bi = blockIdx.x >> 7, o = blockIdx.x & 127, t = threadIdx.x;
  float acc = 0.f;
  for (int q = t; q < 16384; q += 256)
    acc += t2[((size_t)bi*16384 + q)*128 + o];
  __shared__ float sb[256];
  sb[t] = acc; __syncthreads();
  for (int s = 128; s > 0; s >>= 1){ if (t < s) sb[t] += sb[t+s]; __syncthreads(); }
  if (t == 0) p[bi*128 + o] = sb[0] * (1.f/16384.f) + c2b[o];
}

// ---------------- channel attention (tiny 1x1 convs) ----------------
__global__ void k_attn(const float* __restrict__ p, const float* __restrict__ w1,
                       const float* __restrict__ b1, const float* __restrict__ w2,
                       const float* __restrict__ b2, float* __restrict__ attn){
  const int t = threadIdx.x;
  __shared__ float hid[2*CA];
  if (t < 2*CA){
    int bi = t >> 2, j = t & 3;
    float acc = b1[j];
    for (int o = 0; o < 128; ++o) acc += w1[j*128 + o] * p[bi*128 + o];
    hid[t] = fmaxf(acc, 0.f);
  }
  __syncthreads();
  int bi = t >> 7, o = t & 127;
  float acc = b2[o];
  #pragma unroll
  for (int j = 0; j < CA; ++j) acc += w2[o*CA + j] * hid[bi*CA + j];
  attn[t] = sigmoidf_(acc);
}

// ---------------- final: out = x1*skip2 + (t2+c2_b)*attn, NCHW ----------------
__global__ void k_final(const float* __restrict__ x1, const float* __restrict__ t2,
                        const float* __restrict__ c2b, const float* __restrict__ attn,
                        const float* __restrict__ skip2, float* __restrict__ out){
  const int pix = blockIdx.x, c = threadIdx.x;
  const int bi = pix >> 14, y = (pix >> 7) & 127, xx = pix & 127;
  float v = x1[(size_t)pix*128 + c] * skip2[c]
          + (t2[(size_t)pix*128 + c] + c2b[c]) * attn[bi*128 + c];
  out[(((size_t)(bi*128 + c)) << 14) + (y << 7) + xx] = v;
}

// =====================================================================
extern "C" void kernel_launch(void* const* d_in, const int* in_sizes, int n_in,
                              void* d_out, int out_size, void* d_ws, size_t ws_size,
                              hipStream_t stream){
  (void)in_sizes; (void)n_in; (void)out_size; (void)ws_size;
  const float* x        = (const float*)d_in[0];
  const float* ln1_w    = (const float*)d_in[1];
  const float* ln1_b    = (const float*)d_in[2];
  const float* in_proj  = (const float*)d_in[3];
  const float* conv_w   = (const float*)d_in[4];
  const float* conv_b   = (const float*)d_in[5];
  const float* x_proj   = (const float*)d_in[6];
  const float* dt_w     = (const float*)d_in[7];
  const float* dt_b     = (const float*)d_in[8];
  const float* A_log    = (const float*)d_in[9];
  const float* Ds       = (const float*)d_in[10];
  const float* onw      = (const float*)d_in[11];
  const float* onb      = (const float*)d_in[12];
  const float* out_proj = (const float*)d_in[13];
  const float* skip1    = (const float*)d_in[14];
  const float* skip2    = (const float*)d_in[15];
  const float* ln2_w    = (const float*)d_in[16];
  const float* ln2_b    = (const float*)d_in[17];
  const float* c1_w     = (const float*)d_in[18];
  const float* c1_b     = (const float*)d_in[19];
  const float* c2_w     = (const float*)d_in[20];
  const float* c2_b     = (const float*)d_in[21];
  const float* ca1_w    = (const float*)d_in[22];
  const float* ca1_b    = (const float*)d_in[23];
  const float* ca2_w    = (const float*)d_in[24];
  const float* ca2_b    = (const float*)d_in[25];
  float* out = (float*)d_out;

  char* ws = (char*)d_ws;
  _Float16* WB_IN  = (_Float16*)(ws + O_WB_IN);
  _Float16* WB_OUT = (_Float16*)(ws + O_WB_OUT);
  _Float16* WB_XP  = (_Float16*)(ws + O_WB_XP);
  _Float16* WB_C1  = (_Float16*)(ws + O_WB_C1);
  _Float16* WB_C2  = (_Float16*)(ws + O_WB_C2);
  _Float16* XN     = (_Float16*)(ws + O_XN);
  float*    XZ     = (float*)   (ws + O_XZ);
  float*    XC     = (float*)   (ws + O_XC);
  _Float16* XS     = (_Float16*)(ws + O_XS);
  float*    XDBL   = (float*)   (ws + O_XDBL);
  float*    DELTA  = (float*)   (ws + O_DELTA);
  float*    YS     = (float*)   (ws + O_YS);
  _Float16* YG     = (_Float16*)(ws + O_YG);
  float*    YP     = (float*)   (ws + O_YP);
  float*    X1     = (float*)   (ws + O_X1);
  _Float16* TN     = (_Float16*)(ws + O_TN);
  _Float16* A1     = (_Float16*)(ws + O_A1);
  float*    T1     = (float*)   (ws + O_T1);
  _Float16* A2     = (_Float16*)(ws + O_A2);
  float*    T2     = (float*)   (ws + O_T2);
  float*    PBUF   = (float*)   (ws + O_P);
  float*    ATTN   = (float*)   (ws + O_ATTN);

  // weight prep
  k_prep_inproj <<<(512*128 + 255)/256, 256, 0, stream>>>(in_proj, WB_IN);
  k_prep_outproj<<<(128*256 + 255)/256, 256, 0, stream>>>(out_proj, WB_OUT);
  k_prep_xproj  <<<(4*48*256 + 255)/256, 256, 0, stream>>>(x_proj, WB_XP);
  k_prep_c1     <<<(48*K1 + 255)/256, 256, 0, stream>>>(c1_w, WB_C1);
  k_prep_c2     <<<(128*K2P + 255)/256, 256, 0, stream>>>(c2_w, WB_C2);

  // LN1 + shift -> xn (f16)
  k_ln1_shift<<<NPIX, 128, 0, stream>>>(x, ln1_w, ln1_b, XN);

  // xz = xn @ in_proj^T    (M=32768, N=512, K=128)
  k_gemm_wmma<<<dim3(512/16, NPIX/32, 1), 64, 0, stream>>>(XN, WB_IN, XZ, NPIX, 512, 128, 0, 0, 0);

  // depthwise conv + silu
  k_dwconv_silu<<<NPIX, DIN, 0, stream>>>(XZ, conv_w, conv_b, XC);

  // 4-direction z-order gather with shift mask -> xs (f16)
  k_gather_xs<<<dim3(NWIN, KDIR, WTOK), DIN, 0, stream>>>(XC, XS);

  // x_dbl[k] = xs[k] @ x_proj[k]^T   (batched GEMM: M=32768, N=48, K=256)
  k_gemm_wmma<<<dim3(XDBLC/16, NPIX/32, KDIR), 64, 0, stream>>>(
      XS, WB_XP, XDBL, NPIX, XDBLC, DIN,
      (long)NPIX*DIN, (long)XDBLC*DIN, (long)NPIX*XDBLC);

  // delta = softplus(dts @ dt_proj^T + dt_b)
  k_delta<<<dim3(NPIX, KDIR), DIN, 0, stream>>>(XDBL, dt_w, dt_b, DELTA);

  // selective scan
  k_scan<<<dim3(NWIN, KDIR), DIN, 0, stream>>>(XDBL, DELTA, XS, A_log, Ds, YS);

  // combine dirs + window reverse + out_norm + gate -> yg (f16)
  k_combine_gate<<<dim3(NWIN, WTOK), DIN, 0, stream>>>(YS, XZ, onw, onb, YG);

  // yp = yg @ out_proj^T   (M=32768, N=128, K=256)
  k_gemm_wmma<<<dim3(128/16, NPIX/32, 1), 64, 0, stream>>>(YG, WB_OUT, YP, NPIX, 128, DIN, 0, 0, 0);

  // roll back + residual
  k_x1<<<NPIX, 128, 0, stream>>>(x, YP, skip1, X1);

  // LN2 -> tn (f16)
  k_ln2<<<NPIX, 128, 0, stream>>>(X1, ln2_w, ln2_b, TN);

  // conv1 as im2col + WMMA GEMM (M=32768, N=48, K=1152)
  k_im2col1<<<dim3(NPIX, 9), 128, 0, stream>>>(TN, A1);
  k_gemm_wmma<<<dim3(C3P/16, NPIX/32, 1), 64, 0, stream>>>(A1, WB_C1, T1, NPIX, C3P, K1, 0, 0, 0);

  // gelu + conv2 as im2col + WMMA GEMM (M=32768, N=128, K=384)
  k_gelu_im2col2<<<dim3(NPIX, 9), C3P, 0, stream>>>(T1, c1_b, A2);
  k_gemm_wmma<<<dim3(128/16, NPIX/32, 1), 64, 0, stream>>>(A2, WB_C2, T2, NPIX, 128, K2P, 0, 0, 0);

  // channel attention
  k_pool<<<256, 256, 0, stream>>>(T2, c2_b, PBUF);
  k_attn<<<1, 256, 0, stream>>>(PBUF, ca1_w, ca1_b, ca2_w, ca2_b, ATTN);

  // final residual, NCHW output
  k_final<<<NPIX, 128, 0, stream>>>(X1, T2, c2_b, ATTN, skip2, out);
}
